// NeuralNet_21612275433794
// MI455X (gfx1250) — matrix operand order repaired and checked by reading, not compiled, not run on hardware
//
#include <hip/hip_runtime.h>
#include <hip/hip_bf16.h>
#include <stdint.h>

// ---------------------------------------------------------------------------
// SIREN MLP inference on gfx1250 (MI455X), f16 WMMA with f32 accumulation.
//   enc(12, pad->32) -> 128 -> 128 -> 128 -> 128 -> 1, sin activations.
// A = weights (M = neurons), B = activations (N = points): D gives each lane
// 8 contiguous f16 outputs (one b128 store), bias loads straight into C.
// Each layer runs as a pure-WMMA accumulate phase (8 live accumulators, no
// interleaved LDS stores -> partial dscnt waits, prefetchable A-fragments)
// followed by a batched activate+store phase (one WMMA->VALU hazard/layer).
// ---------------------------------------------------------------------------

typedef __attribute__((ext_vector_type(16))) _Float16     v16h;
typedef __attribute__((ext_vector_type(8)))  float        v8f;
typedef __attribute__((ext_vector_type(2)))  _Float16     f16x2;
typedef __attribute__((ext_vector_type(4)))  unsigned int u32x4;

#define NEURONS   128
#define KIN_PAD   32            // 12 real enc features, zero-padded to one K=32 step
#define PI_F      3.14159265358979323846f

#define H_ROW     136           // halves per activation row (128 + 8 pad)
#define H_ROWD    68            // dwords per activation row
#define WH_ROW    136           // halves per hidden-weight row
#define WH_ROWD   68
#define WIN_ROW   40            // halves per W_in row (32 + 8 pad)
#define WIN_ROWD  20

// ---- LDS layout (bytes) ----------------------------------------------------
#define OFF_WIN   0                          // 128 x 40 f16        = 10240
#define OFF_WH    (OFF_WIN  + 10240)         // 3 x 128 x 136 f16   = 104448
#define OFF_WOUT  (OFF_WH   + 104448)        // 128 f32             = 512
#define OFF_BIN   (OFF_WOUT + 512)           // 128 f32             = 512
#define OFF_BH    (OFF_BIN  + 512)           // 384 f32             = 1536
#define OFF_BOUT  (OFF_BH   + 1536)          // 1 f32 (pad 16)      = 16
#define OFF_H     (OFF_BOUT + 16)            // 8 waves x 16 x 136 f16 = 34816
#define LDS_BYTES (OFF_H    + 34816)         // = 152080

union F16frag { v16h h; u32x4 q[2]; };

// A-fragment 16x32 f16 (ISA 7.12.2): lanes<16 row M=lane%16, K={0..7,16..23};
// lanes>=16 same M, K={8..15,24..31}. row = weight row [m][k] as dwords.
__device__ __forceinline__ v16h lds_load_A16x32(const uint32_t* rowChunk, int lane) {
  const int kb = (lane & 16) ? 4 : 0;
  F16frag r;
  r.q[0] = *(const u32x4*)(rowChunk + kb);       // VGPR0..3
  r.q[1] = *(const u32x4*)(rowChunk + 8 + kb);   // VGPR4..7
  return r.h;
}

// B-fragment 32x16 f16: lanes<16 col N=lane%16, K={0..7,16..23}; lanes>=16, K={8..15,24..31}.
// row = activation row [point][k] as dwords; packed K pairs are contiguous.
__device__ __forceinline__ v16h lds_load_B32x16(const uint32_t* rowChunk, int lane) {
  const int kb = (lane & 16) ? 4 : 0;
  F16frag r;
  r.q[0] = *(const u32x4*)(rowChunk + kb);
  r.q[1] = *(const u32x4*)(rowChunk + 8 + kb);
  return r.h;
}

// sin-activate an accumulator and store 8 contiguous f16 with one b128 store.
__device__ __forceinline__ void act_store_b128(_Float16* dst, v8f c) {
  union { u32x4 q; f16x2 hh[4]; } pk;
  #pragma unroll
  for (int i = 0; i < 4; ++i) {
    f16x2 tmp;
    tmp[0] = (_Float16)__sinf(c[2 * i]);
    tmp[1] = (_Float16)__sinf(c[2 * i + 1]);
    pk.hh[i] = tmp;
  }
  *(u32x4*)dst = pk.q;
}

__global__ void __launch_bounds__(256)
siren_mlp_wmma(const float* __restrict__ x,
               const float* __restrict__ W_in,  const float* __restrict__ b_in,
               const float* __restrict__ W_h,   const float* __restrict__ b_h,
               const float* __restrict__ W_out, const float* __restrict__ b_out,
               float* __restrict__ out, int n)
{
  extern __shared__ char smem[];
  _Float16* sWin  = (_Float16*)(smem + OFF_WIN);
  _Float16* sWh   = (_Float16*)(smem + OFF_WH);
  float*    sWout = (float*)   (smem + OFF_WOUT);
  float*    sBin  = (float*)   (smem + OFF_BIN);
  float*    sBh   = (float*)   (smem + OFF_BH);
  float*    sBout = (float*)   (smem + OFF_BOUT);
  _Float16* sH    = (_Float16*)(smem + OFF_H);

  const int t = threadIdx.x;

  // ---- stage all weights to LDS (f32 -> f16, padded rows), once per block --
  for (int i = t; i < NEURONS * KIN_PAD; i += 256) {
    int nr = i >> 5, k = i & 31;
    sWin[nr * WIN_ROW + k] = (k < 12) ? (_Float16)W_in[nr * 12 + k] : (_Float16)0.0f;
  }
  for (int i = t; i < 3 * NEURONS * NEURONS; i += 256) {
    int row = i >> 7, k = i & 127;
    sWh[row * WH_ROW + k] = (_Float16)W_h[i];
  }
  for (int i = t; i < NEURONS; i += 256) { sWout[i] = W_out[i]; sBin[i] = b_in[i]; }
  for (int i = t; i < 3 * NEURONS; i += 256) sBh[i] = b_h[i];
  if (t == 0) sBout[0] = b_out[0];
  __syncthreads();

  const int wave = t >> 5;
  const int lane = t & 31;
  const int sub  = lane & 15;               // A: neuron subrow; B/D: point column
  const int mofs = (lane & 16) ? 8 : 0;     // C/D layout: VGPR r -> M = r + mofs
  const bool hiK = (lane >= 16);
  // Branchless cos/sin: cos(x) = sin(x + pi/2). hiK lanes produce sin terms.
  const float phase = hiK ? 0.0f : (0.5f * PI_F);

  _Float16* Hw  = sH + wave * (16 * H_ROW); // wave-private 16 x 128(+pad) tile
  uint32_t* Hwd = (uint32_t*)Hw;            // H_ROWD dwords per row

  for (int base = blockIdx.x * 128 + wave * 16; base < n; base += gridDim.x * 128) {
    // ---- positional encoding: enc[m][2j]=cos, [2j+1]=sin; pad K 12..31 = 0 --
    const int p  = base + sub;
    const float xv = (p < n) ? x[p] : 0.0f;
    #pragma unroll
    for (int j = 0; j < 6; ++j) {
      const float ang = xv * (PI_F * (float)(1 << j)) + phase;
      Hw[sub * H_ROW + 2 * j + (hiK ? 1 : 0)] = (_Float16)__sinf(ang);
    }
    {
      uint32_t* rowd = Hwd + sub * H_ROWD;
      const int d0 = 6 + (hiK ? 5 : 0);     // dwords 6..15 = K 12..31
      #pragma unroll
      for (int i = 0; i < 5; ++i) rowd[d0 + i] = 0u;
    }

    // ---- input layer: W_in[16x32 A-tiles] x enc[32x16 B], 8 M-tiles --------
    {
      const v16h bEnc = lds_load_B32x16(Hwd + sub * H_ROWD, lane);
      const uint32_t* Wd = (const uint32_t*)sWin;
      v8f acc[8];
      #pragma unroll
      for (int nt = 0; nt < 8; ++nt) {                       // pure-WMMA phase
        v8f c = *(const v8f*)(sBin + nt * 16 + mofs);
        const v16h aW = lds_load_A16x32(Wd + (nt * 16 + sub) * WIN_ROWD, lane);
        acc[nt] = __builtin_amdgcn_wmma_f32_16x16x32_f16(false, aW, false, bEnc,
                                                         (short)0, c, false, false);
      }
      #pragma unroll
      for (int nt = 0; nt < 8; ++nt)                         // store phase
        act_store_b128(Hw + sub * H_ROW + nt * 16 + mofs, acc[nt]);
    }

    // ---- 3 hidden layers: W[16x32 A] x H[32x16 B], 4 K-steps x 8 M-tiles ---
    #pragma unroll
    for (int L = 0; L < 3; ++L) {
      const uint32_t* Wd = (const uint32_t*)(sWh + L * NEURONS * WH_ROW);
      const float* bL = sBh + L * NEURONS;
      const v16h hb0 = lds_load_B32x16(Hwd + sub * H_ROWD +  0, lane);
      const v16h hb1 = lds_load_B32x16(Hwd + sub * H_ROWD + 16, lane);
      const v16h hb2 = lds_load_B32x16(Hwd + sub * H_ROWD + 32, lane);
      const v16h hb3 = lds_load_B32x16(Hwd + sub * H_ROWD + 48, lane);
      v8f acc[8];
      #pragma unroll
      for (int nt = 0; nt < 8; ++nt) {                       // pure-WMMA phase
        v8f c = *(const v8f*)(bL + nt * 16 + mofs);
        const uint32_t* wrow = Wd + (nt * 16 + sub) * WH_ROWD;
        c = __builtin_amdgcn_wmma_f32_16x16x32_f16(false,
              lds_load_A16x32(wrow +  0, lane), false, hb0, (short)0, c, false, false);
        c = __builtin_amdgcn_wmma_f32_16x16x32_f16(false,
              lds_load_A16x32(wrow + 16, lane), false, hb1, (short)0, c, false, false);
        c = __builtin_amdgcn_wmma_f32_16x16x32_f16(false,
              lds_load_A16x32(wrow + 32, lane), false, hb2, (short)0, c, false, false);
        c = __builtin_amdgcn_wmma_f32_16x16x32_f16(false,
              lds_load_A16x32(wrow + 48, lane), false, hb3, (short)0, c, false, false);
        acc[nt] = c;
      }
      #pragma unroll
      for (int nt = 0; nt < 8; ++nt)                         // store phase
        act_store_b128(Hw + sub * H_ROW + nt * 16 + mofs, acc[nt]);
    }

    // ---- output layer (N=1): split-K dot over LDS, 2 lanes per point -------
    {
      const int khalf = lane >> 4;                   // 0: K 0..63, 1: K 64..127
      float s = 0.0f;
      const f16x2* hv = (const f16x2*)(Hw + sub * H_ROW + khalf * 64);
      const float* wo = sWout + khalf * 64;
      #pragma unroll 8
      for (int i = 0; i < 32; ++i) {
        const f16x2 pk = hv[i];
        s += (float)pk[0] * wo[2 * i] + (float)pk[1] * wo[2 * i + 1];
      }
      s += __shfl_xor(s, 16);
      if (khalf == 0 && p < n) out[p] = s + sBout[0];
    }
  }
}

extern "C" void kernel_launch(void* const* d_in, const int* in_sizes, int n_in,
                              void* d_out, int out_size, void* d_ws, size_t ws_size,
                              hipStream_t stream) {
  const float* x     = (const float*)d_in[0];
  const float* W_in  = (const float*)d_in[1];
  const float* b_in  = (const float*)d_in[2];
  const float* W_h   = (const float*)d_in[3];
  const float* b_h   = (const float*)d_in[4];
  const float* W_out = (const float*)d_in[5];
  const float* b_out = (const float*)d_in[6];
  float* out = (float*)d_out;
  const int n = in_sizes[0];

  (void)d_ws; (void)ws_size; (void)n_in; (void)out_size;

  // Allow >64KB dynamic LDS (host-side attribute; not a stream operation).
  hipFuncSetAttribute((const void*)siren_mlp_wmma,
                      hipFuncAttributeMaxDynamicSharedMemorySize, LDS_BYTES);

  const int tiles = (n + 127) / 128;          // 128 points per block-iteration
  const int grid  = tiles < 1296 ? tiles : 1296;
  siren_mlp_wmma<<<grid, 256, LDS_BYTES, stream>>>(
      x, W_in, b_in, W_h, b_h, W_out, b_out, out, n);
}
